// CRF_1649267441994
// MI455X (gfx1250) — matrix operand order repaired and checked by reading, not compiled, not run on hardware
//
#include <hip/hip_runtime.h>
#include <hip/hip_fp16.h>

// ---------------------------------------------------------------------------
// CRF log-domain marginals on gfx1250 (MI455X).
//
// alpha[i,j] = s[i,j] + LSE_k(alpha[i-1,k] + T[k,j])
// beta [i,j] = s[i,j] + LSE_k(T[j,k] + beta[i+1,k])
// out        = alpha + beta - s - Z
//
// Strategy: exp-domain rescaled mat-vec per step, executed with split-f16
// WMMA (hi+lo planes of exp(T), hi+lo planes of the exp'd state vector,
// f32 accumulation) so accuracy stays near-f32 while the matrix op runs on
// v_wmma_f32_16x16x32_f16. Sequential dependency handled by persistent
// kernels with a device-wide atomic grid barrier per step (32 WGs x 64 thr,
// guaranteed co-resident). exp(T) is pre-swizzled into this code's contiguous-k A-fragment
// lane layout so the hot loop is pure 32B fragment loads from L2.
// ---------------------------------------------------------------------------

#define N_LEN   8192
#define C_CLS   1024
#define NWG     32          // workgroups per persistent kernel
#define TPB     64          // 2 wave32 waves; each wave owns one 16-col tile
#define WG_COLS 32          // columns of the state vector owned per WG
#define NKC     32          // K-chunks of 32 over C=1024

typedef __attribute__((ext_vector_type(16))) _Float16 v16h;
typedef __attribute__((ext_vector_type(8)))  float    v8f;

// ---------------------------------------------------------------- barriers
__device__ __forceinline__ void grid_barrier(unsigned* ctr, unsigned target) {
  __threadfence();
  __syncthreads();
  if (threadIdx.x == 0) {
    __hip_atomic_fetch_add(ctr, 1u, __ATOMIC_ACQ_REL, __HIP_MEMORY_SCOPE_AGENT);
    while (__hip_atomic_load(ctr, __ATOMIC_ACQUIRE, __HIP_MEMORY_SCOPE_AGENT) < target)
      __builtin_amdgcn_s_sleep(1);
  }
  __syncthreads();
}

__device__ __forceinline__ float wg_reduce_max(float v, float* red) {
  red[threadIdx.x] = v; __syncthreads();
  for (int s = TPB / 2; s > 0; s >>= 1) {
    if (threadIdx.x < (unsigned)s)
      red[threadIdx.x] = fmaxf(red[threadIdx.x], red[threadIdx.x + s]);
    __syncthreads();
  }
  float r = red[0]; __syncthreads();
  return r;
}

__device__ __forceinline__ float wg_reduce_sum(float v, float* red) {
  red[threadIdx.x] = v; __syncthreads();
  for (int s = TPB / 2; s > 0; s >>= 1) {
    if (threadIdx.x < (unsigned)s)
      red[threadIdx.x] += red[threadIdx.x + s];
    __syncthreads();
  }
  float r = red[0]; __syncthreads();
  return r;
}

// ------------------------------------------------------- fragment pre-swizzle
// A-fragment layout (16-bit A 16x32; contiguous k order used by this code):
//   lane < 16 : M = lane,    halves = K {0..15}
//   lane >= 16: M = lane-16, halves = K {16..31}
// Fragment array index: frag[((jt*NKC + kc)*32 + lane)*16 + hh]
// dir 0 (alpha): A[m][k] = exp(T[k, jbase+m])   (uses columns of T)
// dir 1 (beta) : A[m][k] = exp(T[jbase+m, k])   (uses rows of T)
__global__ void crf_prep(const float* __restrict__ T,
                         _Float16* __restrict__ aHi, _Float16* __restrict__ aLo,
                         _Float16* __restrict__ bHi, _Float16* __restrict__ bLo) {
  unsigned idx  = blockIdx.x * blockDim.x + threadIdx.x;   // 0 .. 2*2^20-1
  unsigned dir  = idx >> 20;
  unsigned e    = idx & 0xFFFFFu;
  unsigned jt   = e >> 14;          // 64 j-tiles
  unsigned kc   = (e >> 9) & 31u;   // 32 k-chunks
  unsigned lane = (e >> 4) & 31u;
  unsigned hh   = e & 15u;
  unsigned M = lane & 15u;
  unsigned K = hh + ((lane >> 4) << 4);
  unsigned j = jt * 16u + M;
  unsigned k = kc * 32u + K;
  float t  = (dir == 0u) ? T[k * C_CLS + j] : T[j * C_CLS + k];
  float ev = __expf(t);                         // ~[e^-5, e^5]: safe in f16
  _Float16 hi = (_Float16)ev;
  _Float16 lo = (_Float16)(ev - (float)hi);     // residual plane
  if (dir == 0u) { aHi[e] = hi; aLo[e] = lo; }
  else           { bHi[e] = hi; bLo[e] = lo; }
}

// --------------------------------------------------- split-f16 WMMA mat-vec
// Computes u[jt*16 .. jt*16+15] = sum_k v[k] * E[tileRow, k] with
// E ~ Ehi+Elo (pre-swizzled), v ~ vh+vl (in LDS). Result tile D has every
// column equal to u; lanes 0/16 carry the 16 values in their 8 accumulators.
__device__ __forceinline__ v8f lse_matvec(const _Float16* __restrict__ fragHi,
                                          const _Float16* __restrict__ fragLo,
                                          int jt,
                                          const _Float16* shVh,
                                          const _Float16* shVl) {
  const int lane    = threadIdx.x & 31;
  const int laneSel = (lane >> 4) << 4;   // B operand: k-offset 0 or 16
  v8f acc = {};
  const _Float16* pa = fragHi + ((size_t)(jt * NKC) * 32 + lane) * 16;
  const _Float16* pl = fragLo + ((size_t)(jt * NKC) * 32 + lane) * 16;
#pragma unroll 4
  for (int kc = 0; kc < NKC; ++kc) {
    __builtin_prefetch(pa + 512, 0, 1);   // next k-chunk fragment (L2)
    v16h ah = *(const v16h*)pa;
    v16h al = *(const v16h*)pl;
    const int kk = kc * 32 + laneSel;
    v16h bh = *(const v16h*)(shVh + kk);
    v16h bl = *(const v16h*)(shVl + kk);
    acc = __builtin_amdgcn_wmma_f32_16x16x32_f16(false, ah, false, bh,
                                                 (short)0, acc, false, false);
    acc = __builtin_amdgcn_wmma_f32_16x16x32_f16(false, ah, false, bl,
                                                 (short)0, acc, false, false);
    acc = __builtin_amdgcn_wmma_f32_16x16x32_f16(false, al, false, bh,
                                                 (short)0, acc, false, false);
    pa += 512;   // 32 lanes * 16 halves per fragment
    pl += 512;
  }
  return acc;
}

// Build vh/vl = exp(row - m) split into LDS; returns m. 64 thr x 16 elems.
__device__ __forceinline__ float build_v(const float* __restrict__ row,
                                         _Float16* shVh, _Float16* shVl,
                                         float* red) {
  float av[16];
  float lm = -3.4e38f;
#pragma unroll
  for (int t = 0; t < 16; ++t) {
    av[t] = row[threadIdx.x * 16 + t];
    lm = fmaxf(lm, av[t]);
  }
  const float m = wg_reduce_max(lm, red);
#pragma unroll
  for (int t = 0; t < 16; ++t) {
    const int k = threadIdx.x * 16 + t;
    const float v = __expf(av[t] - m);          // in (0, 1]
    const _Float16 vh = (_Float16)v;
    shVh[k] = vh;
    shVl[k] = (_Float16)(v - (float)vh);
  }
  __syncthreads();
  return m;
}

// ------------------------------------------------------------- alpha pass
__global__ void crf_alpha(const float* __restrict__ scores,
                          const _Float16* __restrict__ fragHi,
                          const _Float16* __restrict__ fragLo,
                          float* __restrict__ out,        // alpha rows
                          unsigned* __restrict__ ctr) {
  __shared__ __attribute__((aligned(32))) _Float16 shVh[C_CLS];
  __shared__ __attribute__((aligned(32))) _Float16 shVl[C_CLS];
  __shared__ float red[TPB];
  const int tid  = threadIdx.x;
  const int g    = blockIdx.x;
  const int wave = tid >> 5;
  const int lane = tid & 31;
  const int jt   = g * 2 + wave;            // this wave's 16-column tile
  unsigned bar = 0;

  if (tid < WG_COLS) {                       // alpha[0] = scores[0]
    const int j = g * WG_COLS + tid;
    out[j] = scores[j];
  }
  grid_barrier(ctr, (++bar) * NWG);

  for (int i = 1; i < N_LEN; ++i) {
    const float m = build_v(out + (size_t)(i - 1) * C_CLS, shVh, shVl, red);
    const v8f acc = lse_matvec(fragHi, fragLo, jt, shVh, shVl);
    if (lane == 0 || lane == 16) {
      const int jb = jt * 16 + ((lane >> 4) << 3);
      const float* srow = scores + (size_t)i * C_CLS;
      float* orow = out + (size_t)i * C_CLS;
#pragma unroll
      for (int r = 0; r < 8; ++r)
        orow[jb + r] = srow[jb + r] + m + __logf(acc[r]);
    }
    grid_barrier(ctr, (++bar) * NWG);
  }
}

// ----------------------------------------------- beta pass (fused marginals)
__global__ void crf_beta(const float* __restrict__ scores,
                         const _Float16* __restrict__ fragHi,
                         const _Float16* __restrict__ fragLo,
                         float* __restrict__ out,        // alpha in, marg out
                         float* __restrict__ bbuf,       // 2*C double buffer
                         unsigned* __restrict__ ctr) {
  __shared__ __attribute__((aligned(32))) _Float16 shVh[C_CLS];
  __shared__ __attribute__((aligned(32))) _Float16 shVl[C_CLS];
  __shared__ float red[TPB];
  const int tid  = threadIdx.x;
  const int g    = blockIdx.x;
  const int wave = tid >> 5;
  const int lane = tid & 31;
  const int jt   = g * 2 + wave;
  unsigned bar = 0;

  // Z = LSE(alpha[N-1]) computed redundantly per WG (trivial cost)
  const float* alast = out + (size_t)(N_LEN - 1) * C_CLS;
  float av[16];
  float lm = -3.4e38f;
#pragma unroll
  for (int t = 0; t < 16; ++t) {
    av[t] = alast[tid * 16 + t];
    lm = fmaxf(lm, av[t]);
  }
  const float mz = wg_reduce_max(lm, red);
  float ls = 0.f;
#pragma unroll
  for (int t = 0; t < 16; ++t) ls += __expf(av[t] - mz);
  const float Z = mz + __logf(wg_reduce_sum(ls, red));

  if (tid < WG_COLS) {                       // beta[N-1] = scores[N-1]
    const int j = g * WG_COLS + tid;
    const size_t o = (size_t)(N_LEN - 1) * C_CLS + j;
    bbuf[((N_LEN - 1) & 1) * C_CLS + j] = scores[o];
    out[o] = alast[j] - Z;                   // alpha + beta - s - Z, beta==s
  }
  grid_barrier(ctr, (++bar) * NWG);

  for (int i = N_LEN - 2; i >= 0; --i) {
    const float m = build_v(bbuf + (size_t)((i + 1) & 1) * C_CLS, shVh, shVl, red);
    const v8f acc = lse_matvec(fragHi, fragLo, jt, shVh, shVl);
    if (lane == 0 || lane == 16) {
      const int jb = jt * 16 + ((lane >> 4) << 3);
      const float* srow = scores + (size_t)i * C_CLS;
      float* orow = out + (size_t)i * C_CLS;
      float* brow = bbuf + (size_t)(i & 1) * C_CLS;
#pragma unroll
      for (int r = 0; r < 8; ++r) {
        const int j = jb + r;
        const float b = srow[j] + m + __logf(acc[r]);
        brow[j] = b;
        orow[j] = orow[j] + b - srow[j] - Z;   // orow held alpha[i,j]
      }
    }
    grid_barrier(ctr, (++bar) * NWG);
  }
}

// --------------------------------------------------------------------- host
extern "C" void kernel_launch(void* const* d_in, const int* in_sizes, int n_in,
                              void* d_out, int out_size, void* d_ws, size_t ws_size,
                              hipStream_t stream) {
  const float* scores = (const float*)d_in[0];   // [8192,1024] f32
  const float* T      = (const float*)d_in[1];   // [1024,1024] f32
  float* out = (float*)d_out;                    // [8192,1024] f32

  // workspace layout: 4 x 2MB fragment planes, beta double buffer, counters
  char* ws = (char*)d_ws;
  const size_t PLANE = (size_t)C_CLS * C_CLS * sizeof(_Float16);  // 2 MB
  _Float16* aHi = (_Float16*)(ws);
  _Float16* aLo = (_Float16*)(ws + PLANE);
  _Float16* bHi = (_Float16*)(ws + 2 * PLANE);
  _Float16* bLo = (_Float16*)(ws + 3 * PLANE);
  float*    bbuf = (float*)(ws + 4 * PLANE);                      // 8 KB
  unsigned* ctrs = (unsigned*)(ws + 4 * PLANE + 2 * C_CLS * sizeof(float));

  (void)hipMemsetAsync(ctrs, 0, 256, stream);    // reset grid-barrier counters

  crf_prep<<<(2u << 20) / 256u, 256, 0, stream>>>(T, aHi, aLo, bHi, bLo);
  crf_alpha<<<NWG, TPB, 0, stream>>>(scores, aHi, aLo, out, ctrs + 0);
  crf_beta <<<NWG, TPB, 0, stream>>>(scores, bHi, bLo, out, bbuf, ctrs + 64);

  (void)in_sizes; (void)n_in; (void)out_size; (void)ws_size;
}
